// Non_local_Block_62362925137961
// MI455X (gfx1250) — matrix operand order repaired and checked by reading, not compiled, not run on hardware
//
#include <hip/hip_runtime.h>

// ---------------- types ----------------
typedef __bf16 bf16;
typedef __attribute__((ext_vector_type(16))) __bf16 bf16x16;
typedef __attribute__((ext_vector_type(8)))  __bf16 bf16x8;
typedef __attribute__((ext_vector_type(8)))  float  v8f;

#define NB    4
#define CIN   256
#define CO    128
#define NPIX  4096   // 64*64

// ---------------- WMMA fragment loaders (ISA 7.12.2 layouts) ----------------
// A operand, 16x32 bf16, src row-major [rows][ldk]; lane row = row (caller passes q0+l16).
__device__ __forceinline__ bf16x16 load_frag_a(const bf16* __restrict__ base, int ldk,
                                               int row, int k0, int hf) {
  const bf16* p = base + (size_t)row * ldk + k0;
  bf16x8 lo = *(const bf16x8*)(p + hf * 8);
  bf16x8 hi = *(const bf16x8*)(p + 16 + hf * 8);
  bf16x16 r;
#pragma unroll
  for (int i = 0; i < 8; ++i) { r[i] = lo[i]; r[i + 8] = hi[i]; }
  return r;
}

// B operand, 32x16 bf16, NT: B column n stored as row n of src [cols][ldk].
__device__ __forceinline__ bf16x16 load_frag_b(const bf16* __restrict__ base, int ldk,
                                               int col, int k0, int hf) {
  const bf16* p = base + (size_t)col * ldk + k0 + hf * 8;
  bf16x8 lo = *(const bf16x8*)(p);
  bf16x8 hi = *(const bf16x8*)(p + 16);
  bf16x16 r;
#pragma unroll
  for (int i = 0; i < 8; ++i) { r[i] = lo[i]; r[i + 8] = hi[i]; }
  return r;
}

__device__ __forceinline__ v8f wmma_bf16(bf16x16 a, bf16x16 b, v8f c) {
  return __builtin_amdgcn_wmma_f32_16x16x32_bf16(false, a, false, b, (short)0, c,
                                                 false, false);
}

// gfx1250 async global->LDS copy, 16B per lane, tracked with ASYNCcnt.
// VDST = per-lane LDS byte offset (workgroup-relative), VADDR = 64-bit global addr.
__device__ __forceinline__ unsigned lds_off(const void* p) {
  return (unsigned)(unsigned long long)p;   // generic LDS addr: low 32 bits = offset
}
__device__ __forceinline__ void async_cp16(unsigned lds_byte_off, const void* gsrc) {
  asm volatile("global_load_async_to_lds_b128 %0, %1, off"
               :: "v"(lds_byte_off), "v"(gsrc) : "memory");
}

// ---------------- stage 0a: weights f32 -> bf16 ----------------
__global__ void k_wconv(const float* __restrict__ wt, const float* __restrict__ wp,
                        const float* __restrict__ wg, const float* __restrict__ ww,
                        bf16* ot, bf16* op, bf16* og, bf16* ow) {
  int i = blockIdx.x * 256 + threadIdx.x;
  if (i < CO * CIN) {
    ot[i] = (bf16)wt[i];
    op[i] = (bf16)wp[i];
    og[i] = (bf16)wg[i];
    ow[i] = (bf16)ww[i];
  }
}

// ---------------- stage 0b: x [B][CIN][N] f32 -> xT [B][N][CIN] bf16 ----------------
__global__ void k_xpose(const float* __restrict__ x, bf16* __restrict__ xT) {
  __shared__ float t[16][17];
  int p0 = blockIdx.x * 16, c0 = blockIdx.y * 16, b = blockIdx.z;
  int tx = threadIdx.x, ty = threadIdx.y;
  t[ty][tx] = x[((size_t)b * CIN + c0 + ty) * NPIX + p0 + tx];
  __syncthreads();
  xT[((size_t)b * NPIX + p0 + ty) * CIN + c0 + tx] = (bf16)t[tx][ty];
}

// ---------------- stage 1: fused theta/phi/g projections ----------------
__global__ void k_proj(const bf16* __restrict__ xT,
                       const bf16* __restrict__ wth, const bf16* __restrict__ wph,
                       const bf16* __restrict__ wg,
                       const float* __restrict__ bth, const float* __restrict__ bph,
                       const float* __restrict__ bg,
                       bf16* __restrict__ th, bf16* __restrict__ ph,
                       bf16* __restrict__ gT) {
  int wave = threadIdx.x >> 5, lane = threadIdx.x & 31;
  int l16 = lane & 15, hf = lane >> 4;
  int tile = blockIdx.x * 8 + wave;           // B * 256 * 8 = 8192 tiles
  int b = tile >> 11;
  int rem = tile & 2047;
  int pt = rem >> 3, ot = rem & 7;
  int q0 = pt * 16, o0 = ot * 16;
  const bf16* xb = xT + (size_t)b * NPIX * CIN;

  v8f a0 = {}, a1 = {}, a2 = {};
#pragma unroll
  for (int kk = 0; kk < 8; ++kk) {            // K = CIN = 256
    bf16x16 ax = load_frag_a(xb, CIN, q0 + l16, kk * 32, hf);
    a0 = wmma_bf16(ax, load_frag_b(wth, CIN, o0 + l16, kk * 32, hf), a0);
    a1 = wmma_bf16(ax, load_frag_b(wph, CIN, o0 + l16, kk * 32, hf), a1);
    a2 = wmma_bf16(ax, load_frag_b(wg,  CIN, o0 + l16, kk * 32, hf), a2);
  }
  float bt = bth[o0 + l16], bp = bph[o0 + l16], bgv = bg[o0 + l16];
  bf16* thb = th + (size_t)b * NPIX * CO;
  bf16* phb = ph + (size_t)b * NPIX * CO;
  bf16* gTb = gT + (size_t)b * CO * NPIX;
#pragma unroll
  for (int r = 0; r < 8; ++r) {
    int p = q0 + r + 8 * hf;
    thb[(size_t)p * CO + o0 + l16] = (bf16)(a0[r] + bt);
    phb[(size_t)p * CO + o0 + l16] = (bf16)(a1[r] + bp);
    gTb[(size_t)(o0 + l16) * NPIX + p] = (bf16)(a2[r] + bgv);
  }
}

// ---------------- stage 2: fused flash attention ----------------
// 8 waves/block share one m-block of phi/g, staged in LDS by async copies
// (double-buffered). Each wave owns 16 query rows; S never touches HBM.
__global__ void __launch_bounds__(256) k_attn(const bf16* __restrict__ th,
                                              const bf16* __restrict__ ph,
                                              const bf16* __restrict__ gT,
                                              bf16* __restrict__ y) {
  __shared__ __attribute__((aligned(16))) bf16 sphi[2][32 * CO];   // 2 x 8KB
  __shared__ __attribute__((aligned(16))) bf16 sg[2][CO * 32];     // 2 x 8KB
  __shared__ __attribute__((aligned(16))) bf16 pbuf[8][16 * 32];   // 8KB P staging
  int tid = threadIdx.x;
  int wave = tid >> 5, lane = tid & 31;
  int l16 = lane & 15, hf = lane >> 4;
  int b = blockIdx.x >> 5;                       // 32 blocks per batch
  int q0 = (((blockIdx.x & 31) << 3) + wave) * 16;
  const bf16* thb = th + (size_t)b * NPIX * CO;
  const bf16* phb = ph + (size_t)b * NPIX * CO;
  const bf16* gTb = gT + (size_t)b * CO * NPIX;
  bf16* pl = &pbuf[wave][0];

  // cooperative async staging of one m-block (32 cols): 4 x b128 per thread
  unsigned phi_dst0 = lds_off(&sphi[0][0]) + tid * 16;
  unsigned phi_dst1 = lds_off(&sphi[1][0]) + tid * 16;
  int grow = tid >> 1, gh = tid & 1;
  unsigned g_dst0 = lds_off(&sg[0][0]) + grow * 64 + gh * 32;
  unsigned g_dst1 = lds_off(&sg[1][0]) + grow * 64 + gh * 32;
  const char* gsrc_row = (const char*)(gTb + (size_t)grow * NPIX) + gh * 32;

#define ISSUE_COPIES(bufsel, m0_)                                               \
  do {                                                                          \
    const char* ps = (const char*)(phb + (size_t)(m0_) * CO) + tid * 16;        \
    unsigned pd = (bufsel) ? phi_dst1 : phi_dst0;                               \
    async_cp16(pd, ps);                                                         \
    async_cp16(pd + 4096, ps + 4096);                                           \
    const char* gs = gsrc_row + (size_t)(m0_) * 2;                              \
    unsigned gd = (bufsel) ? g_dst1 : g_dst0;                                   \
    async_cp16(gd, gs);                                                         \
    async_cp16(gd + 16, gs + 16);                                               \
  } while (0)

  bf16x16 aq[4];                              // theta fragments: resident
#pragma unroll
  for (int kk = 0; kk < 4; ++kk) aq[kk] = load_frag_a(thb, CO, q0 + l16, kk * 32, hf);

  v8f O[8];
  float mrun[8], lrun[8];
#pragma unroll
  for (int c = 0; c < 8; ++c) O[c] = v8f{};
#pragma unroll
  for (int r = 0; r < 8; ++r) { mrun[r] = -INFINITY; lrun[r] = 0.f; }

  ISSUE_COPIES(0, 0);

  for (int m0 = 0; m0 < NPIX; m0 += 32) {
    int cur = (m0 >> 5) & 1;
    if (m0 + 32 < NPIX) {
      ISSUE_COPIES(cur ^ 1, m0 + 32);
      asm volatile("s_wait_asynccnt 0x4" ::: "memory");  // current block landed
    } else {
      asm volatile("s_wait_asynccnt 0x0" ::: "memory");
    }
    __syncthreads();                                     // all waves' copies visible
    const bf16* lphi = &sphi[cur][0];
    const bf16* lg   = &sg[cur][0];

    // ---- S = theta x phi^T : two 16x16 tiles, K=CO=128, B from LDS ----
    v8f s0 = {}, s1 = {};
#pragma unroll
    for (int kk = 0; kk < 4; ++kk)
      s0 = wmma_bf16(aq[kk], load_frag_b(lphi, CO, l16, kk * 32, hf), s0);
#pragma unroll
    for (int kk = 0; kk < 4; ++kk)
      s1 = wmma_bf16(aq[kk], load_frag_b(lphi, CO, 16 + l16, kk * 32, hf), s1);

    // ---- online softmax; acc layout: row = r+8*hf (vgpr), col = l16 (lane) ----
    float scale[8];
#pragma unroll
    for (int r = 0; r < 8; ++r) {
      float t = fmaxf(s0[r], s1[r]);
#pragma unroll
      for (int mk = 1; mk < 16; mk <<= 1) t = fmaxf(t, __shfl_xor(t, mk, 32));
      float mn = fmaxf(mrun[r], t);
      float sc = __expf(mrun[r] - mn);
      float e0 = __expf(s0[r] - mn);
      float e1 = __expf(s1[r] - mn);
      float rs = e0 + e1;
#pragma unroll
      for (int mk = 1; mk < 16; mk <<= 1) rs += __shfl_xor(rs, mk, 32);
      lrun[r] = lrun[r] * sc + rs;
      mrun[r] = mn;
      scale[r] = sc;
      pl[(r + 8 * hf) * 32 + l16]      = (bf16)e0;
      pl[(r + 8 * hf) * 32 + 16 + l16] = (bf16)e1;
    }
#pragma unroll
    for (int c = 0; c < 8; ++c)
#pragma unroll
      for (int r = 0; r < 8; ++r) O[c][r] *= scale[r];

    asm volatile("s_wait_dscnt 0x0" ::: "memory");       // per-wave P staging done

    // ---- O += P x g : A from LDS (K=32), B from LDS g block (ld=32) ----
    bf16x16 pa = load_frag_a(pl, 32, l16, 0, hf);
#pragma unroll
    for (int c = 0; c < 8; ++c)
      O[c] = wmma_bf16(pa, load_frag_b(lg, 32, c * 16 + l16, 0, hf), O[c]);

    __syncthreads();                                     // protect bufs before reuse
  }
#undef ISSUE_COPIES

  // ---- epilogue: y[b][p][co] = O / l ----
  float inv[8];
#pragma unroll
  for (int r = 0; r < 8; ++r) inv[r] = 1.0f / lrun[r];
  bf16* yb = y + ((size_t)b * NPIX + q0) * CO;
#pragma unroll
  for (int c = 0; c < 8; ++c)
#pragma unroll
    for (int r = 0; r < 8; ++r)
      yb[(size_t)(r + 8 * hf) * CO + c * 16 + l16] = (bf16)(O[c][r] * inv[r]);
}

// ---------------- stage 3: z = y x wW^T + bW + x, store [B][CIN][N] f32 ----------------
__global__ void k_outconv(const bf16* __restrict__ y, const bf16* __restrict__ wW,
                          const float* __restrict__ bW, const float* __restrict__ x,
                          float* __restrict__ out) {
  int wave = threadIdx.x >> 5, lane = threadIdx.x & 31;
  int l16 = lane & 15, hf = lane >> 4;
  int tile = blockIdx.x * 8 + wave;           // B * 256 * 16 = 16384 tiles
  int b = tile >> 12;
  int rem = tile & 4095;
  int pt = rem >> 4, ct = rem & 15;
  int q0 = pt * 16, c0 = ct * 16;
  const bf16* yb = y + (size_t)b * NPIX * CO;

  v8f acc = {};
#pragma unroll
  for (int kk = 0; kk < 4; ++kk)              // K = CO = 128
    acc = wmma_bf16(load_frag_a(yb, CO, q0 + l16, kk * 32, hf),
                    load_frag_b(wW, CO, c0 + l16, kk * 32, hf), acc);

  float bias = bW[c0 + l16];
#pragma unroll
  for (int r = 0; r < 8; ++r) {
    int p = q0 + r + 8 * hf;
    size_t idx = ((size_t)b * CIN + c0 + l16) * NPIX + p;
    out[idx] = acc[r] + bias + x[idx];
  }
}

// ---------------- host launcher ----------------
extern "C" void kernel_launch(void* const* d_in, const int* in_sizes, int n_in,
                              void* d_out, int out_size, void* d_ws, size_t ws_size,
                              hipStream_t stream) {
  (void)in_sizes; (void)n_in; (void)out_size; (void)ws_size;
  const float* x       = (const float*)d_in[0];
  const float* w_theta = (const float*)d_in[1];
  const float* b_theta = (const float*)d_in[2];
  const float* w_phi   = (const float*)d_in[3];
  const float* b_phi   = (const float*)d_in[4];
  const float* w_g     = (const float*)d_in[5];
  const float* b_g     = (const float*)d_in[6];
  const float* w_W     = (const float*)d_in[7];
  const float* b_W     = (const float*)d_in[8];

  char* ws = (char*)d_ws;
  bf16* xT  = (bf16*)(ws);                    // 8,388,608 B
  bf16* th  = (bf16*)(ws + 8388608);          // 4,194,304 B each
  bf16* ph  = (bf16*)(ws + 12582912);
  bf16* gT  = (bf16*)(ws + 16777216);
  bf16* yb  = (bf16*)(ws + 20971520);
  bf16* wth = (bf16*)(ws + 25165824);         // 65,536 B each
  bf16* wph = (bf16*)(ws + 25231360);
  bf16* wgv = (bf16*)(ws + 25296896);
  bf16* wWb = (bf16*)(ws + 25362432);

  k_wconv<<<128, 256, 0, stream>>>(w_theta, w_phi, w_g, w_W, wth, wph, wgv, wWb);

  dim3 tg(NPIX / 16, CIN / 16, NB), tb(16, 16);
  k_xpose<<<tg, tb, 0, stream>>>(x, xT);

  k_proj<<<1024, 256, 0, stream>>>(xT, wth, wph, wgv, b_theta, b_phi, b_g, th, ph, gT);

  k_attn<<<128, 256, 0, stream>>>(th, ph, gT, yb);

  k_outconv<<<2048, 256, 0, stream>>>(yb, wWb, b_W, x, (float*)d_out);
}
